// StochMultiScaleTransformer_44787918962967
// MI455X (gfx1250) — matrix operand order repaired and checked by reading, not compiled, not run on hardware
//
#include <hip/hip_runtime.h>
#include <hip/hip_fp16.h>

// ---------------------------------------------------------------------------
// StochMultiScaleTransformer forward, CDNA5 (gfx1250) WMMA implementation.
// B=16 S=64 HH=16 L=4 ACT=16 FDIM=32 D=512 NH=8 DH=64 MLP=2048 NL=4
// All big GEMMs in f16 WMMA (v_wmma_f32_16x16x32_f16) with f32 accumulate.
// MROW=2: each wave computes 2 M-subtiles sharing each B fragment (2 wmma
// per 32B weight-fragment load, half the weight re-reads).
// Algebraic reductions: k_blk/v_blk pushed through the linear S-mean so the
// per-block wk/wv GEMMs shrink from 65536 rows to 1024 rows; mu/logvar are
// never materialized (epilogues atomically accumulate S-mean and KLD sums).
// ---------------------------------------------------------------------------

typedef __attribute__((ext_vector_type(8)))  _Float16 v8h;
typedef __attribute__((ext_vector_type(16))) _Float16 v16h;
typedef __attribute__((ext_vector_type(8)))  float    v8f;

#define DEV __device__ __forceinline__

DEV float geluf(float x){ return 0.5f*x*(1.0f + erff(x*0.70710678118654752f)); }
DEV void  atomAddF(float* p, float v){
  __hip_atomic_fetch_add(p, v, __ATOMIC_RELAXED, __HIP_MEMORY_SCOPE_AGENT);
}

// ---------------------------------------------------------------------------
// small utility kernels
// ---------------------------------------------------------------------------
__global__ void k_zero(float* p, int n){
  int i = blockIdx.x*256 + threadIdx.x;
  if(i < n) p[i] = 0.f;
}

__global__ void k_scale_f16(const float* src, _Float16* dst, float s, int n){
  int i = blockIdx.x*256 + threadIdx.x;
  if(i < n) dst[i] = (_Float16)(src[i]*s);
}

// WT[n*K+k] = (f16) W[k*N+n]   (N-major transpose for WMMA B fragments)
__global__ void k_w_to_f16T(const float* W, _Float16* WT, int K, int N){
  int i = blockIdx.x*256 + threadIdx.x;
  if(i < K*N){
    int n = i / K, k = i - n*K;
    WT[i] = (_Float16)W[(size_t)k*N + n];
  }
}

// ---------------------------------------------------------------------------
// generic WMMA GEMM: C[M,N] = A[M,K] (f16, row-major) @ WT[N,K] (f16) + bias
// block: 256 threads = 8 waves, 16*MROW M-rows, 8*TPW*16 N-cols.
// NFULL=true: N is an exact multiple of the block width (no bound checks).
// ---------------------------------------------------------------------------
enum { EPI_PLAIN=0, EPI_ADD, EPI_GELU, EPI_LN_GELU, EPI_LN2, EPI_MU, EPI_LV };

struct GemmP {
  const _Float16* A; const _Float16* WT; const float* bias;
  int M, N, K, ldd;
  float* outF32; _Float16* outF16; _Float16* outF16b;
  const float* g0; const float* c0; const float* g1; const float* c1;
  float outScale;
  float* accPtr;     // KLD partial sums target
  float* atomicDst;  // kmean accumulation target (EPI_MU)
};

template<int TPW, int EPI, int MROW, bool NFULL>
__global__ __launch_bounds__(256) void k_gemm(GemmP p){
  extern __shared__ char smemRaw[];
  const int tid  = threadIdx.x;
  const int lane = tid & 31;
  const int wid  = tid >> 5;
  const int m0   = blockIdx.y * (16*MROW);
  const int bw   = 8*TPW*16;
  const int nb0  = blockIdx.x * bw;
  const int nw0  = nb0 + wid*TPW*16;

  v8f acc[MROW][TPW];
  #pragma unroll
  for(int mi=0;mi<MROW;mi++)
    #pragma unroll
    for(int t=0;t<TPW;t++)
      #pragma unroll
      for(int j=0;j<8;j++) acc[mi][t][j] = 0.f;

  const int arow = m0 + (lane & 15);
  const _Float16* Abase0 = p.A + (size_t)arow * p.K + 16*(lane>>4);
  // per-wave B fragment base: column (nw0 + lane&15), K offset 16*(lane>>4)
  const _Float16* Bbase = p.WT + (size_t)(nw0 + (lane&15)) * p.K + 16*(lane>>4);
  const size_t Btile = (size_t)16 * p.K;   // stride between the wave's tiles
  const size_t Arow16 = (size_t)16 * p.K;  // stride between M-subtiles

  for(int k0=0; k0<p.K; k0+=32){
    if(k0 + 32 < p.K) __builtin_prefetch(Abase0 + k0 + 128, 0, 1);
    v16h a[MROW];
    #pragma unroll
    for(int mi=0;mi<MROW;mi++){
      const _Float16* Ap = Abase0 + mi*Arow16 + k0;
      v8h alo = *(const v8h*)(Ap);
      v8h ahi = *(const v8h*)(Ap + 8);
      #pragma unroll
      for(int i=0;i<8;i++){ a[mi][i]=alo[i]; a[mi][i+8]=ahi[i]; }
    }
    #pragma unroll
    for(int t=0;t<TPW;t++){
      if(NFULL || (nw0 + t*16) < p.N){
        const _Float16* Bp = Bbase + (size_t)t*Btile + k0;
        v8h blo = *(const v8h*)(Bp);
        v8h bhi = *(const v8h*)(Bp + 8);
        v16h b;
        #pragma unroll
        for(int i=0;i<8;i++){ b[i]=blo[i]; b[i+8]=bhi[i]; }
        #pragma unroll
        for(int mi=0;mi<MROW;mi++)
          acc[mi][t] = __builtin_amdgcn_wmma_f32_16x16x32_f16(
              false, a[mi], false, b, (short)0, acc[mi][t], false, false);
      }
    }
  }

  // C fragment mapping (per subtile): lane<16 -> rows +j, lane>=16 -> rows +8+j;
  // col = n0 + (lane&15)
  if(EPI==EPI_PLAIN || EPI==EPI_ADD || EPI==EPI_GELU){
    #pragma unroll
    for(int t=0;t<TPW;t++){
      int ncol = nw0 + t*16 + (lane & 15);
      if(!NFULL && ncol >= p.N) continue;
      float bv = p.bias ? p.bias[ncol] : 0.f;
      #pragma unroll
      for(int mi=0;mi<MROW;mi++){
        #pragma unroll
        for(int j=0;j<8;j++){
          int r = m0 + mi*16 + j + 8*(lane>>4);
          float v = acc[mi][t][j] + bv;
          if(EPI==EPI_GELU) v = geluf(v);
          v *= p.outScale;
          size_t idx = (size_t)r * p.ldd + ncol;
          if(EPI==EPI_ADD)      p.outF32[idx] += v;
          else {
            if(p.outF32) p.outF32[idx] = v;
            if(p.outF16) p.outF16[idx] = (_Float16)v;
          }
        }
      }
    }
  }

  if(EPI==EPI_LN_GELU || EPI==EPI_LN2){
    // block spans the full row width (grid.x == 1): layernorm over N
    constexpr int ROWS = 16*MROW;
    constexpr int TPR  = 256/ROWS;          // threads per row in reduction
    float* slab  = (float*)smemRaw;         // ROWS * N
    float* red   = slab + (size_t)ROWS*p.N; // 256*2
    float* stats = red + 512;               // ROWS*2
    #pragma unroll
    for(int t=0;t<TPW;t++){
      int ncol = nw0 + t*16 + (lane & 15);
      float bv = p.bias ? p.bias[ncol] : 0.f;
      #pragma unroll
      for(int mi=0;mi<MROW;mi++){
        #pragma unroll
        for(int j=0;j<8;j++){
          int rl = mi*16 + j + 8*(lane>>4);
          slab[rl*p.N + ncol] = acc[mi][t][j] + bv;
        }
      }
    }
    __syncthreads();
    int rl = tid / TPR, sub = tid % TPR;
    float s=0.f, s2=0.f;
    for(int c=sub; c<p.N; c+=TPR){ float x = slab[rl*p.N + c]; s+=x; s2+=x*x; }
    red[tid*2+0]=s; red[tid*2+1]=s2;
    __syncthreads();
    if(sub==0){
      float S=0.f, S2=0.f;
      for(int i=0;i<TPR;i++){ S += red[(rl*TPR+i)*2]; S2 += red[(rl*TPR+i)*2+1]; }
      float mean = S/(float)p.N;
      float var  = S2/(float)p.N - mean*mean;
      stats[rl*2]   = mean;
      stats[rl*2+1] = rsqrtf(var + 1e-5f);
    }
    __syncthreads();
    for(int i=tid; i<ROWS*p.N; i+=256){
      int r = i / p.N, c = i - r*p.N;
      float xn = (slab[i] - stats[r*2]) * stats[r*2+1];
      size_t idx = (size_t)(m0+r) * p.ldd + c;
      if(EPI==EPI_LN_GELU){
        p.outF16[idx] = (_Float16)geluf(xn*p.g0[c] + p.c0[c]);
      } else {
        p.outF16 [idx] = (_Float16)(xn*p.g0[c] + p.c0[c]);
        p.outF16b[idx] = (_Float16)(xn*p.g1[c] + p.c1[c]);
      }
    }
  }

  if(EPI==EPI_MU || EPI==EPI_LV){
    float local = 0.f;
    #pragma unroll
    for(int t=0;t<TPW;t++){
      int ncol = nw0 + t*16 + (lane & 15);
      float bv = p.bias[ncol];
      #pragma unroll
      for(int mi=0;mi<MROW;mi++){
        #pragma unroll
        for(int j=0;j<8;j++){
          int r = m0 + mi*16 + j + 8*(lane>>4);
          float v = acc[mi][t][j] + bv;
          if(EPI==EPI_MU){
            int g = ((r>>8)<<2) | (r & 3);          // (b*16+hh)*4 + l
            atomAddF(p.atomicDst + (size_t)g*p.N + ncol, v);
            local += v*v;
          } else {
            local += v - __expf(v);
          }
        }
      }
    }
    float* red = (float*)smemRaw;
    red[tid] = local;
    __syncthreads();
    if(tid==0){
      float S=0.f;
      for(int i=0;i<256;i++) S += red[i];
      atomAddF(p.accPtr, S);
    }
  }
}

// ---------------------------------------------------------------------------
// row layernorm (f32 -> f16), one block per row of width N
// ---------------------------------------------------------------------------
__global__ __launch_bounds__(256) void k_ln_rows(const float* x, _Float16* y,
                                                 const float* g, const float* c, int N){
  __shared__ float red[256];
  const int r = blockIdx.x;
  const float* xr = x + (size_t)r*N;
  float s=0.f, s2=0.f;
  for(int i=threadIdx.x;i<N;i+=256){ float v=xr[i]; s+=v; s2+=v*v; }
  red[threadIdx.x]=s; __syncthreads();
  for(int o=128;o>0;o>>=1){ if(threadIdx.x<o) red[threadIdx.x]+=red[threadIdx.x+o]; __syncthreads(); }
  float mean = red[0]/(float)N; __syncthreads();
  red[threadIdx.x]=s2; __syncthreads();
  for(int o=128;o>0;o>>=1){ if(threadIdx.x<o) red[threadIdx.x]+=red[threadIdx.x+o]; __syncthreads(); }
  float var = red[0]/(float)N - mean*mean;
  float inv = rsqrtf(var + 1e-5f);
  for(int i=threadIdx.x;i<N;i+=256)
    y[(size_t)r*N+i] = (_Float16)((xr[i]-mean)*inv*g[i] + c[i]);
}

// ---------------------------------------------------------------------------
// time embedding: emb[b, j<256]=sin(t*f_j), j>=256: cos(t*f_{j-256})
// ---------------------------------------------------------------------------
__global__ void k_temb(const int* t, _Float16* emb){
  int b = blockIdx.x;
  float tv = (float)t[b];
  for(int j=threadIdx.x;j<512;j+=256){
    int jj = j & 255;
    float freq = __expf(-9.210340371976184f * (float)jj / 255.f); // log(10000)
    float a = tv*freq;
    emb[b*512+j] = (_Float16)((j<256) ? __sinf(a) : __cosf(a));
  }
}

// h[r,col] = noisy[r,:16] @ q_projW[:,col] + q_projB[col] + te2[b,col]
__global__ void k_init_h(const float* noisy, const float* qw, const float* qb,
                         const float* te2, float* h){
  int i = blockIdx.x*256 + threadIdx.x;   // 0 .. 256*512-1
  int r = i >> 9, col = i & 511;
  int b = r >> 4;
  float s = qb[col] + te2[b*512+col];
  #pragma unroll
  for(int a=0;a<16;a++) s += noisy[r*16+a]*qw[a*512+col];
  h[i] = s;
}

// ---------------------------------------------------------------------------
// attention core (per batch b): scores->softmax->entropy->scale-fusion->ctx
// q: (256,512) f32 ; kblk: (1024,512) f32 [(b,hk,l) rows] ; vfull: (1024,512)
// ctx16: (16,16,512) f16 with the reference's flat reshape semantics
// ---------------------------------------------------------------------------
__global__ __launch_bounds__(256) void k_attn(const float* q, const float* kblk,
                                              const float* vfull, _Float16* ctx16){
  __shared__ float sAttn[4*8*16*16];  // [l][n][q][k]
  __shared__ float sEnt [4*8*16];     // [l][n][q]
  __shared__ float sEntM[4*16];       // [l][q]
  __shared__ float sW   [4*16];       // [l][q]
  __shared__ float sFus [8*16*16];    // [n][q][k]
  const int b = blockIdx.x, tid = threadIdx.x;
  const float scale = 0.125f;         // DH^-0.5

  for(int idx=tid; idx<512; idx+=256){
    int l = idx>>7, n = (idx>>4)&7, qq = idx&15;
    const float* qp = q + ((size_t)(b*16+qq)*512 + n*64);
    float sc[16]; float mx = -1e30f;
    for(int k=0;k<16;k++){
      const float* kp = kblk + ((size_t)((b*16+k)*4 + l)*512 + n*64);
      float d = 0.f;
      for(int e=0;e<64;e++) d += qp[e]*kp[e];
      sc[k] = d*scale; mx = fmaxf(mx, sc[k]);
    }
    float se=0.f;
    for(int k=0;k<16;k++){ sc[k]=__expf(sc[k]-mx); se+=sc[k]; }
    float inv = 1.f/se, ent = 0.f;
    for(int k=0;k<16;k++){
      float a = sc[k]*inv;
      sAttn[((l*8+n)*16+qq)*16+k] = a;
      ent -= a*__logf(a + 1e-8f);
    }
    sEnt[(l*8+n)*16+qq] = ent;
  }
  __syncthreads();
  if(tid < 64){
    int l = tid>>4, qq = tid&15;
    float s=0.f;
    for(int n=0;n<8;n++) s += sEnt[(l*8+n)*16+qq];
    sEntM[l*16+qq] = (s/8.f)/2.772588722239781f;   // log(16)
  }
  __syncthreads();
  if(tid < 16){
    int qq = tid;
    float mx=-1e30f;
    for(int l=0;l<4;l++) mx = fmaxf(mx, -sEntM[l*16+qq]);
    float e[4]; float s=0.f;
    for(int l=0;l<4;l++){ e[l]=__expf(-sEntM[l*16+qq]-mx); s+=e[l]; }
    for(int l=0;l<4;l++) sW[l*16+qq] = e[l]/s;
  }
  __syncthreads();
  for(int idx=tid; idx<2048; idx+=256){
    int n=idx>>8, qq=(idx>>4)&15, k=idx&15;
    float f=0.f;
    for(int l=0;l<4;l++) f += sW[l*16+qq]*sAttn[((l*8+n)*16+qq)*16+k];
    sFus[idx] = f;
  }
  __syncthreads();
  for(int idx=tid; idx<8192; idx+=256){
    int n=idx>>10, qq=(idx>>6)&15, d=idx&63;
    float c=0.f;
    for(int k=0;k<16;k++){
      float vv = vfull[((size_t)((b*16+k)*4 + 3)*512) + n*64 + d];
      c += sFus[(n*16+qq)*16+k]*vv;
    }
    ctx16[(size_t)b*8192 + idx] = (_Float16)c;   // faithful flat reshape
  }
}

__global__ void k_kld(const float* acc, float* out){
  if(threadIdx.x==0 && blockIdx.x==0)
    out[0] = -0.5f*(512.f + (acc[1]-acc[0])/65536.f);
}

// ---------------------------------------------------------------------------
// host side
// ---------------------------------------------------------------------------
template<int TPW, int EPI, int MROW=1, bool NFULL=true>
static void launch_gemm(const GemmP& p, hipStream_t stream){
  const int bw = 8*TPW*16;
  dim3 grid((unsigned)((p.N + bw - 1)/bw), (unsigned)(p.M/(16*MROW)));
  size_t sh = 0;
  if(EPI==EPI_LN_GELU || EPI==EPI_LN2) sh = ((size_t)16*MROW*p.N + 512 + 64)*sizeof(float);
  else if(EPI==EPI_MU || EPI==EPI_LV)  sh = 256*sizeof(float);
  k_gemm<TPW,EPI,MROW,NFULL><<<grid, dim3(256), sh, stream>>>(p);
}

static GemmP gp(const _Float16* A, const _Float16* WT, const float* bias,
                int M, int N, int K, int ldd){
  GemmP p; p.A=A; p.WT=WT; p.bias=bias; p.M=M; p.N=N; p.K=K; p.ldd=ldd;
  p.outF32=nullptr; p.outF16=nullptr; p.outF16b=nullptr;
  p.g0=nullptr; p.c0=nullptr; p.g1=nullptr; p.c1=nullptr;
  p.outScale=1.f; p.accPtr=nullptr; p.atomicDst=nullptr;
  return p;
}

extern "C" void kernel_launch(void* const* d_in, const int* in_sizes, int n_in,
                              void* d_out, int out_size, void* d_ws, size_t ws_size,
                              hipStream_t stream){
  (void)in_sizes; (void)n_in; (void)out_size; (void)ws_size;
  const float* noisy = (const float*)d_in[0];
  const int*   tval  = (const int*)  d_in[1];
  const float* apxf  = (const float*)d_in[2];
  const float* apxt  = (const float*)d_in[3];
  auto F = [&](int i){ return (const float*)d_in[i]; };
  // params: 4 q_proj_w,5 q_proj_b,6 kv_f_w,7 kv_f_b,8 kv_t_w,9 kv_t_b,
  // 10 kv0_w,11 kv0_b,12 kv_ln_g,13 kv_ln_b,14 kv1_w,15 kv1_b,16 mu_ln_g,
  // 17 mu_ln_b,18 mu_w,19 mu_b,20 lv_ln_g,21 lv_ln_b,22 lv_w,23 lv_b,
  // 24 temb_w,25 temb_b,26 tmlp_w,27 tmlp_b, blocks @28 (16 each), 92..95 out

  char* wsb = (char*)d_ws; size_t off = 0;
  auto alloc = [&](size_t bytes)->char*{
    char* p = wsb + off; off = (off + bytes + 255) & ~(size_t)255; return p;
  };
  const size_t R = 65536;                 // B*HH*S*L rows

  // --- f16 transposed weights -------------------------------------------------
  _Float16* kvfT  = (_Float16*)alloc(512*32*2);
  _Float16* kvtT  = (_Float16*)alloc(512*32*2);
  _Float16* kv0T  = (_Float16*)alloc((size_t)1024*1024*2);
  _Float16* kv1T  = (_Float16*)alloc((size_t)512*1024*2);
  _Float16* muT   = (_Float16*)alloc((size_t)512*512*2);
  _Float16* lvT   = (_Float16*)alloc((size_t)512*512*2);
  _Float16* tembT = (_Float16*)alloc((size_t)512*512*2);
  _Float16* tmlpT = (_Float16*)alloc((size_t)512*512*2);
  _Float16 *wqT[4], *wkT[4], *wvT[4], *woT[4], *ff0T[4], *ff1T[4];
  for(int i=0;i<4;i++){
    wqT[i]=(_Float16*)alloc((size_t)512*512*2);  wkT[i]=(_Float16*)alloc((size_t)512*512*2);
    wvT[i]=(_Float16*)alloc((size_t)512*512*2);  woT[i]=(_Float16*)alloc((size_t)512*512*2);
    ff0T[i]=(_Float16*)alloc((size_t)2048*512*2);
    ff1T[i]=(_Float16*)alloc((size_t)512*2048*2);
  }
  _Float16* outT  = (_Float16*)alloc(16*512*2);

  // --- activations ------------------------------------------------------------
  _Float16* apxf16 = (_Float16*)alloc(R*32*2);
  _Float16* apxt16 = (_Float16*)alloc(R*32*2);
  _Float16* kcat16 = (_Float16*)alloc(R*1024*2);    // reused for kfmu/kflv
  _Float16* h1_16  = (_Float16*)alloc(R*1024*2);
  _Float16* kfmu16 = kcat16;                         // alias (kcat dead after GEMM2)
  _Float16* kflv16 = kcat16 + R*512;
  float*    kmeanF = (float*)   alloc((size_t)1024*512*4);
  _Float16* kmean16= (_Float16*)alloc((size_t)1024*512*2);
  float*    kblkF  = (float*)   alloc((size_t)1024*512*4);
  float*    vfullF = (float*)   alloc((size_t)1024*512*4);
  float*    hF     = (float*)   alloc((size_t)256*512*4);
  _Float16* hn16   = (_Float16*)alloc((size_t)256*512*2);
  float*    qF     = (float*)   alloc((size_t)256*512*4);
  _Float16* ctx16  = (_Float16*)alloc((size_t)256*512*2);
  _Float16* ffm16  = (_Float16*)alloc((size_t)256*2048*2);
  _Float16* emb16  = (_Float16*)alloc(16*512*2);
  _Float16* te1_16 = (_Float16*)alloc(16*512*2);
  float*    te2F   = (float*)   alloc(16*512*4);
  float*    acc    = (float*)   alloc(256);          // [0]=sum mu^2, [1]=sum lv-exp(lv)

  auto tr = [&](const float* W, _Float16* WT, int K, int N){
    int n = K*N;
    k_w_to_f16T<<<dim3((n+255)/256), dim3(256), 0, stream>>>(W, WT, K, N);
  };

  // zero accumulators
  k_zero<<<dim3(2048), dim3(256), 0, stream>>>(kmeanF, 1024*512);
  k_zero<<<dim3(1),    dim3(256), 0, stream>>>(acc, 2);

  // weight conversion / transpose
  tr(F(6),  kvfT, 32, 512);    tr(F(8),  kvtT, 32, 512);
  tr(F(10), kv0T, 1024, 1024); tr(F(14), kv1T, 1024, 512);
  tr(F(18), muT, 512, 512);    tr(F(22), lvT, 512, 512);
  tr(F(24), tembT, 512, 512);  tr(F(26), tmlpT, 512, 512);
  for(int i=0;i<4;i++){
    int bidx = 28 + i*16;
    tr(F(bidx+2),  wqT[i], 512, 512);  tr(F(bidx+4),  wkT[i], 512, 512);
    tr(F(bidx+6),  wvT[i], 512, 512);  tr(F(bidx+8),  woT[i], 512, 512);
    tr(F(bidx+12), ff0T[i], 512, 2048); tr(F(bidx+14), ff1T[i], 2048, 512);
  }
  tr(F(94), outT, 512, 16);

  // activation conversions
  k_scale_f16<<<dim3((int)(R*32/256)), dim3(256), 0, stream>>>(apxf, apxf16, 1.f, (int)(R*32));
  k_scale_f16<<<dim3((int)(R*32/256)), dim3(256), 0, stream>>>(apxt, apxt16, 1.f, (int)(R*32));

  // --- kv encoder path --------------------------------------------------------
  { GemmP p = gp(apxf16, kvfT, F(7),  (int)R, 512, 32, 1024); p.outF16 = kcat16;
    launch_gemm<4,EPI_PLAIN,2>(p, stream); }
  { GemmP p = gp(apxt16, kvtT, F(9),  (int)R, 512, 32, 1024); p.outF16 = kcat16 + 512;
    launch_gemm<4,EPI_PLAIN,2>(p, stream); }
  { GemmP p = gp(kcat16, kv0T, F(11), (int)R, 1024, 1024, 1024);
    p.outF16 = h1_16; p.g0 = F(12); p.c0 = F(13);
    launch_gemm<8,EPI_LN_GELU,2>(p, stream); }
  { GemmP p = gp(h1_16, kv1T, F(15), (int)R, 512, 1024, 512);
    p.outF16 = kfmu16; p.outF16b = kflv16;
    p.g0 = F(16); p.c0 = F(17); p.g1 = F(20); p.c1 = F(21);
    launch_gemm<4,EPI_LN2,2>(p, stream); }
  { GemmP p = gp(kfmu16, muT, F(19), (int)R, 512, 512, 512);
    p.atomicDst = kmeanF; p.accPtr = acc + 0;
    launch_gemm<4,EPI_MU,2>(p, stream); }
  { GemmP p = gp(kflv16, lvT, F(23), (int)R, 512, 512, 512);
    p.accPtr = acc + 1;
    launch_gemm<4,EPI_LV,2>(p, stream); }
  k_scale_f16<<<dim3(2048), dim3(256), 0, stream>>>(kmeanF, kmean16, 1.f/64.f, 1024*512);

  // --- time embedding + h init ------------------------------------------------
  k_temb<<<dim3(16), dim3(256), 0, stream>>>(tval, emb16);
  { GemmP p = gp(emb16, tembT, F(25), 16, 512, 512, 512); p.outF16 = te1_16;
    launch_gemm<4,EPI_GELU,1>(p, stream); }
  { GemmP p = gp(te1_16, tmlpT, F(27), 16, 512, 512, 512); p.outF32 = te2F;
    launch_gemm<4,EPI_PLAIN,1>(p, stream); }
  k_init_h<<<dim3(512), dim3(256), 0, stream>>>(noisy, F(4), F(5), te2F, hF);

  // --- transformer blocks -----------------------------------------------------
  for(int i=0;i<4;i++){
    int bidx = 28 + i*16;
    k_ln_rows<<<dim3(256), dim3(256), 0, stream>>>(hF, hn16, F(bidx+0), F(bidx+1), 512);
    { GemmP p = gp(hn16, wqT[i], F(bidx+3), 256, 512, 512, 512); p.outF32 = qF;
      launch_gemm<4,EPI_PLAIN,2>(p, stream); }
    { GemmP p = gp(kmean16, wkT[i], F(bidx+5), 1024, 512, 512, 512); p.outF32 = kblkF;
      launch_gemm<4,EPI_PLAIN,2>(p, stream); }
    { GemmP p = gp(kmean16, wvT[i], F(bidx+7), 1024, 512, 512, 512); p.outF32 = vfullF;
      p.outScale = 64.f;                         // v_blk = sum_S = 64*(mean@wv + bv)
      launch_gemm<4,EPI_PLAIN,2>(p, stream); }
    k_attn<<<dim3(16), dim3(256), 0, stream>>>(qF, kblkF, vfullF, ctx16);
    { GemmP p = gp(ctx16, woT[i], F(bidx+9), 256, 512, 512, 512); p.outF32 = hF;
      launch_gemm<4,EPI_ADD,2>(p, stream); }
    k_ln_rows<<<dim3(256), dim3(256), 0, stream>>>(hF, hn16, F(bidx+10), F(bidx+11), 512);
    { GemmP p = gp(hn16, ff0T[i], F(bidx+13), 256, 2048, 512, 2048); p.outF16 = ffm16;
      launch_gemm<8,EPI_GELU,2>(p, stream); }
    { GemmP p = gp(ffm16, ff1T[i], F(bidx+15), 256, 512, 2048, 512); p.outF32 = hF;
      launch_gemm<4,EPI_ADD,2>(p, stream); }
  }

  // --- output head ------------------------------------------------------------
  k_ln_rows<<<dim3(256), dim3(256), 0, stream>>>(hF, hn16, F(92), F(93), 512);
  { GemmP p = gp(hn16, outT, F(95), 256, 16, 512, 16); p.outF32 = (float*)d_out;
    launch_gemm<1,EPI_PLAIN,1,false>(p, stream); }
  k_kld<<<dim3(1), dim3(64), 0, stream>>>(acc, (float*)d_out + 4096);
}
